// lstm_source_14903536517340
// MI455X (gfx1250) — matrix operand order repaired and checked by reading, not compiled, not run on hardware
//
#include <hip/hip_runtime.h>

typedef __attribute__((ext_vector_type(16))) _Float16 v16h;
typedef __attribute__((ext_vector_type(8)))  _Float16 v8h;
typedef __attribute__((ext_vector_type(8)))  float    v8f;

#define WAVES_PER_BLOCK 8

__device__ __forceinline__ float sigmoidf_fast(float x) {
    return 1.0f / (1.0f + __expf(-x));
}

// ---------------------------------------------------------------------------
// Cooperative async stage of a contiguous 16xK f16 A-slab into LDS.
// Uses gfx1250 GLOBAL_LOAD_ASYNC_TO_LDS_B128 (ASYNCcnt-tracked, no VGPR data)
// then s_wait_asynccnt + workgroup barrier. 256 threads; elems % 2048 == 0.
// ---------------------------------------------------------------------------
__device__ __forceinline__ void stage_a_async(const _Float16* __restrict__ gA,
                                              _Float16* lds, int elems) {
    const int nvec = elems >> 3;                    // 16-byte chunks
    const unsigned lbase = (unsigned)(uintptr_t)lds; // low 32 bits = LDS offset
    for (int i = threadIdx.x; i < nvec; i += 256) {
        const _Float16* src = gA + (size_t)i * 8;
        unsigned dst = lbase + (unsigned)i * 16u;
        asm volatile("global_load_async_to_lds_b128 %0, %1, off"
                     :: "v"(dst), "v"(src) : "memory");
    }
#if __has_builtin(__builtin_amdgcn_s_wait_asynccnt)
    __builtin_amdgcn_s_wait_asynccnt(0);
#else
    asm volatile("s_wait_asynccnt 0x0" ::: "memory");
#endif
    __syncthreads();
}

// ---------------------------------------------------------------------------
// Pack a row-major f32 weight matrix B[K][N] into WMMA B-fragment order (f16).
//   lanes 0-15 : N = ntile*16 + lane,     K = kb*32 + 0..7, 16..23  (v0..v7)
//   lanes 16-31: N = ntile*16 + lane-16,  K = kb*32 + 8..15, 24..31 (v0..v7)
// dst[((ntile*(K/32)+kb)*32 + lane)*16 + e]  <- src[k*N + n]
// ---------------------------------------------------------------------------
__global__ __launch_bounds__(256) void pack_b_wmma(const float* __restrict__ src,
                                                   _Float16* __restrict__ dst,
                                                   int K, int N) {
    int tid = blockIdx.x * blockDim.x + threadIdx.x;
    int total = K * N;
    if (tid >= total) return;
    int e     = tid & 15;
    int lane  = (tid >> 4) & 31;
    int kbn   = K >> 5;
    int kb    = (tid >> 9) % kbn;
    int ntile = (tid >> 9) / kbn;
    int n = ntile * 16 + (lane & 15);
    int k = kb * 32 + ((lane >> 4) << 3) + (e & 7) + ((e >> 3) << 4);
    dst[tid] = (_Float16)src[(size_t)k * N + n];
}

__global__ __launch_bounds__(256) void f32_to_f16_kernel(const float* __restrict__ src,
                                                         _Float16* __restrict__ dst, int n) {
    int i = blockIdx.x * blockDim.x + threadIdx.x;
    if (i < n) dst[i] = (_Float16)src[i];
}

__global__ __launch_bounds__(256) void copy_f32_kernel(const float* __restrict__ src,
                                                       float* __restrict__ dst, int n) {
    int i = blockIdx.x * blockDim.x + threadIdx.x;
    if (i < n) dst[i] = src[i];
}

// ---------------------------------------------------------------------------
// WMMA GEMM:  C[M][N] = A[M][K] (f16 row-major) @ Bp (packed f16) + bias[N];
// optional ReLU; optional f16 shadow copy of C.
// Block = 8 waves sharing one 16-row mtile (requires (N/64) % 8 == 0 and
// grid exactly (M/16)*(N/64)/8).  A slab async-staged to LDS once, then the
// hot loop reads A via ds_load_b128 and streams B tiles from L2.
// ---------------------------------------------------------------------------
__global__ __launch_bounds__(256) void gemm_f16_wmma(const _Float16* __restrict__ A,
                                                     const _Float16* __restrict__ Bp,
                                                     const float* __restrict__ bias,
                                                     float* __restrict__ C,
                                                     _Float16* __restrict__ Ch,
                                                     int M, int N, int K, int relu) {
    __shared__ _Float16 Atile[16 * 1024];          // up to K=1024 -> 32KB

    const int wid  = threadIdx.x >> 5;
    const int lane = threadIdx.x & 31;
    const int ngrp = N >> 6;                       // groups of 4 N-tiles
    const int gid  = blockIdx.x * WAVES_PER_BLOCK + wid;
    const int mtile = gid / ngrp;                  // uniform across the block
    const int ng    = gid % ngrp;
    const int kbn   = K >> 5;

    stage_a_async(A + (size_t)mtile * 16 * K, Atile, 16 * K);

    v8f acc[4] = {};
    const int khalf = (lane >> 4) << 3;            // 0 or 8
    const _Float16* arow = Atile + (size_t)(lane & 15) * K + khalf;

    for (int kb = 0; kb < kbn; ++kb) {
        union { v16h v; v8h h[2]; } a;
        a.h[0] = *(const v8h*)(arow + kb * 32);        // K+0..7   (or 8..15)
        a.h[1] = *(const v8h*)(arow + kb * 32 + 16);   // K+16..23 (or 24..31)
        #pragma unroll
        for (int g = 0; g < 4; ++g) {
            const _Float16* bp =
                Bp + (((size_t)(ng * 4 + g) * kbn + kb) * 32 + lane) * 16;
            v16h b = *(const v16h*)bp;
            acc[g] = __builtin_amdgcn_wmma_f32_16x16x32_f16(
                false, a.v, false, b, (short)0, acc[g], false, false);
        }
    }

    const int rbase = mtile * 16 + ((lane >> 4) << 3);
    #pragma unroll
    for (int g = 0; g < 4; ++g) {
        const int n = (ng * 4 + g) * 16 + (lane & 15);
        const float bv = bias ? bias[n] : 0.0f;
        #pragma unroll
        for (int r = 0; r < 8; ++r) {
            const int mr = rbase + r;
            float v = acc[g][r] + bv;
            if (relu) v = fmaxf(v, 0.0f);
            C[(size_t)mr * N + n] = v;
            if (Ch) Ch[(size_t)mr * N + n] = (_Float16)v;
        }
    }
}

// ---------------------------------------------------------------------------
// One fused recurrent step:  z = xW[:,t,:] + h@U ; gates ; c,h update.
// H=1024, B=32 fixed.  Grid = 16 blocks x 8 waves = 128 waves:
// mtile in {0,1} (uniform per block) x 64 h-column tiles.  The wave's 4
// accumulators are the 4 gates, so gate fusion happens in-register.
// h slab (16x1024) is async-staged to LDS and shared by the block's 8 waves.
// ---------------------------------------------------------------------------
__global__ __launch_bounds__(256) void lstm_step_wmma(const _Float16* __restrict__ hin,
                                                      _Float16* __restrict__ hout,
                                                      const _Float16* __restrict__ Up,
                                                      const float* __restrict__ xW,
                                                      float* __restrict__ c_state,
                                                      float* __restrict__ h_state,
                                                      _Float16* __restrict__ seq,
                                                      int t, int T) {
    __shared__ _Float16 Htile[16 * 1024];          // 32KB

    const int wid  = threadIdx.x >> 5;
    const int lane = threadIdx.x & 31;
    const int gid  = blockIdx.x * WAVES_PER_BLOCK + wid;   // 0..127
    const int mtile = gid >> 6;    // 0..1, uniform per block
    const int nt    = gid & 63;    // h-column tile

    stage_a_async(hin + (size_t)mtile * 16 * 1024, Htile, 16 * 1024);

    v8f acc[4] = {};
    const int khalf = (lane >> 4) << 3;
    const _Float16* arow = Htile + (size_t)(lane & 15) * 1024 + khalf;

    for (int kb = 0; kb < 32; ++kb) {
        union { v16h v; v8h h[2]; } a;
        a.h[0] = *(const v8h*)(arow + kb * 32);
        a.h[1] = *(const v8h*)(arow + kb * 32 + 16);
        #pragma unroll
        for (int g = 0; g < 4; ++g) {
            const int ntile = (g << 6) + nt;               // gate g columns
            const _Float16* bp = Up + (((size_t)ntile * 32 + kb) * 32 + lane) * 16;
            v16h b = *(const v16h*)bp;
            acc[g] = __builtin_amdgcn_wmma_f32_16x16x32_f16(
                false, a.v, false, b, (short)0, acc[g], false, false);
        }
    }

    const int n     = nt * 16 + (lane & 15);
    const int rbase = mtile * 16 + ((lane >> 4) << 3);
    #pragma unroll
    for (int r = 0; r < 8; ++r) {
        const int b = rbase + r;
        const size_t xbase = ((size_t)b * T + t) * 4096;
        const float zi = acc[0][r] + xW[xbase + n];
        const float zf = acc[1][r] + xW[xbase + 1024 + n];
        const float zg = acc[2][r] + xW[xbase + 2048 + n];
        const float zo = acc[3][r] + xW[xbase + 3072 + n];
        const float ig = sigmoidf_fast(zi);
        const float fg = sigmoidf_fast(zf);
        const float gg = fmaxf(zg, 0.0f);
        const float og = sigmoidf_fast(zo);
        const size_t sidx = (size_t)b * 1024 + n;
        const float cn = fg * c_state[sidx] + ig * gg;
        c_state[sidx] = cn;
        const float hv = og * fmaxf(cn, 0.0f);
        h_state[sidx] = hv;                                  // final-h slot
        hout[sidx] = (_Float16)hv;                           // next step's A
        seq[((size_t)b * T + t) * 1024 + n] = (_Float16)hv;  // layer output
    }
}

// ---------------------------------------------------------------------------
extern "C" void kernel_launch(void* const* d_in, const int* in_sizes, int n_in,
                              void* d_out, int out_size, void* d_ws, size_t ws_size,
                              hipStream_t stream) {
    (void)in_sizes; (void)n_in; (void)out_size; (void)ws_size;
    constexpr int Bsz = 32, T = 128, E = 512, H = 1024;
    constexpr int BT  = Bsz * T;       // 4096
    constexpr int G4H = 4 * H;         // 4096

    const float* x   = (const float*)d_in[0];
    const float* h0  = (const float*)d_in[1];
    const float* c0  = (const float*)d_in[2];
    const float* W[4]  = {(const float*)d_in[3], (const float*)d_in[6],
                          (const float*)d_in[9], (const float*)d_in[12]};
    const float* U[4]  = {(const float*)d_in[4], (const float*)d_in[7],
                          (const float*)d_in[10], (const float*)d_in[13]};
    const float* bb[4] = {(const float*)d_in[5], (const float*)d_in[8],
                          (const float*)d_in[11], (const float*)d_in[14]};
    const float* Wd1 = (const float*)d_in[15];
    const float* bd1 = (const float*)d_in[16];
    const float* Wd2 = (const float*)d_in[17];
    const float* bd2 = (const float*)d_in[18];

    float* out      = (float*)d_out;                 // (B,T,E)
    float* h_final  = out + (size_t)BT * E;          // (B,H)
    float* c_state  = h_final + (size_t)Bsz * H;     // (B,H) — live c state

    // ---- workspace carve-out (all offsets 256B aligned) ----
    char* ws = (char*)d_ws;
    size_t off = 0;
    auto take = [&](size_t bytes) -> char* {
        char* p = ws + off;
        off += (bytes + 255) & ~(size_t)255;
        return p;
    };
    _Float16* Wp[4];
    Wp[0] = (_Float16*)take((size_t)E * G4H * 2);
    for (int l = 1; l < 4; ++l) Wp[l] = (_Float16*)take((size_t)H * G4H * 2);
    _Float16* Upk[4];
    for (int l = 0; l < 4; ++l) Upk[l] = (_Float16*)take((size_t)H * G4H * 2);
    _Float16* Wd1p  = (_Float16*)take((size_t)H * H * 2);
    _Float16* Wd2p  = (_Float16*)take((size_t)H * E * 2);
    _Float16* x16   = (_Float16*)take((size_t)BT * E * 2);
    _Float16* seq16 = (_Float16*)take((size_t)BT * H * 2);
    _Float16* d116  = (_Float16*)take((size_t)BT * H * 2);
    _Float16* hA    = (_Float16*)take((size_t)Bsz * H * 2);
    _Float16* hB    = (_Float16*)take((size_t)Bsz * H * 2);
    float*    xWbuf = (float*)   take((size_t)BT * G4H * 4);

    auto cdiv = [](int a, int b) { return (a + b - 1) / b; };

    // ---- pack all weights into WMMA fragment order (f16) ----
    for (int l = 0; l < 4; ++l) {
        const int Kin = (l == 0) ? E : H;
        pack_b_wmma<<<cdiv(Kin * G4H, 256), 256, 0, stream>>>(W[l], Wp[l], Kin, G4H);
        pack_b_wmma<<<cdiv(H * G4H, 256), 256, 0, stream>>>(U[l], Upk[l], H, G4H);
    }
    pack_b_wmma<<<cdiv(H * H, 256), 256, 0, stream>>>(Wd1, Wd1p, H, H);
    pack_b_wmma<<<cdiv(H * E, 256), 256, 0, stream>>>(Wd2, Wd2p, H, E);

    // ---- activations / state init ----
    f32_to_f16_kernel<<<cdiv(BT * E, 256), 256, 0, stream>>>(x, x16, BT * E);
    f32_to_f16_kernel<<<cdiv(Bsz * H, 256), 256, 0, stream>>>(h0, hA, Bsz * H);
    copy_f32_kernel<<<cdiv(Bsz * H, 256), 256, 0, stream>>>(c0, c_state, Bsz * H);

    // ---- 4 LSTM layers: bulk xW GEMM, then 128 fused recurrent steps ----
    int step = 0;
    for (int l = 0; l < 4; ++l) {
        const _Float16* Ain = (l == 0) ? x16 : seq16;
        const int Kin = (l == 0) ? E : H;
        const int groups = (BT / 16) * (G4H / 64);       // divisible by 8
        gemm_f16_wmma<<<groups / WAVES_PER_BLOCK, 256, 0, stream>>>(
            Ain, Wp[l], bb[l], xWbuf, (_Float16*)nullptr, BT, G4H, Kin, 0);
        for (int t = 0; t < T; ++t) {
            const _Float16* hin = (step & 1) ? hB : hA;
            _Float16*      hout = (step & 1) ? hA : hB;
            lstm_step_wmma<<<16, 256, 0, stream>>>(hin, hout, Upk[l], xWbuf,
                                                   c_state, h_final, seq16, t, T);
            ++step;
        }
    }

    // ---- dense head: relu(seq @ Wd1 + bd1) @ Wd2 + bd2 ----
    {
        const int groups = (BT / 16) * (H / 64);
        gemm_f16_wmma<<<groups / WAVES_PER_BLOCK, 256, 0, stream>>>(
            seq16, Wd1p, bd1, xWbuf, d116, BT, H, H, 1);
    }
    {
        const int groups = (BT / 16) * (E / 64);
        gemm_f16_wmma<<<groups / WAVES_PER_BLOCK, 256, 0, stream>>>(
            d116, Wd2p, bd2, out, (_Float16*)nullptr, BT, E, H, 0);
    }
}
